// EmbeddingToIndex_45243185496126
// MI455X (gfx1250) — matrix operand order repaired and checked by reading, not compiled, not run on hardware
//
#include <hip/hip_runtime.h>
#include <hip/hip_bf16.h>

typedef __attribute__((ext_vector_type(16))) __bf16 v16bf;
typedef __attribute__((ext_vector_type(8)))  float  v8f;
typedef __attribute__((ext_vector_type(4)))  float  v4f;

#define MROWS   16384   // MB*SEQ = 8*2048
#define NTOK    8192
#define EMB     64
#define NTILES  (NTOK / 16)       // 512
#define NHALF   (NTILES / 2)      // 256 iterations, two N-halves in flight

// ---------------------------------------------------------------------------
// Kernel 1: split codebook into bf16 hi/lo planes + compute ||w_n||^2 (f32).
// One wave per codebook row (2 elements per lane).
// ---------------------------------------------------------------------------
__global__ __launch_bounds__(256) void vq_prep_kernel(
    const float* __restrict__ W, __bf16* __restrict__ whi,
    __bf16* __restrict__ wlo, float* __restrict__ wsq) {
  const int lane = threadIdx.x & 31;
  const int wave = threadIdx.x >> 5;
  const int n    = blockIdx.x * 8 + wave;           // 1024 blocks * 8 waves
  const float* wr = W + (size_t)n * EMB;
  const int k = lane * 2;
  float e0 = wr[k], e1 = wr[k + 1];
  __bf16 h0 = (__bf16)e0, h1 = (__bf16)e1;
  __bf16 l0 = (__bf16)(e0 - (float)h0), l1 = (__bf16)(e1 - (float)h1);
  whi[(size_t)n * EMB + k] = h0;  whi[(size_t)n * EMB + k + 1] = h1;
  wlo[(size_t)n * EMB + k] = l0;  wlo[(size_t)n * EMB + k + 1] = l1;
  float sq = e0 * e0 + e1 * e1;
  #pragma unroll
  for (int off = 16; off >= 1; off >>= 1) sq += __shfl_xor(sq, off, 32);
  if (lane == 0) wsq[n] = sq;
}

// ---------------------------------------------------------------------------
// Kernel 2: fused bf16x3 WMMA GEMM + running argmin.
// Block = 256 threads = 8 waves = 4 M-tiles x 2 N-halves. Block covers 64 rows.
// B tiles staged through LDS and shared by all waves; prefetch double-buffered
// in registers. Three independent WMMA accumulator chains break the C-matrix
// RAW dependency so the matrix pipe can stay full.
// ---------------------------------------------------------------------------
__global__ __launch_bounds__(256) void vq_argmin_kernel(
    const float* __restrict__ X, const __bf16* __restrict__ whi,
    const __bf16* __restrict__ wlo, const float* __restrict__ wsq,
    float* __restrict__ out) {
  // [chunk: h0/hi, h0/lo, h1/hi, h1/lo][row 0..15][k 0..63]
  __shared__ __align__(16) __bf16 sB[4][16][EMB];   // 8 KB
  __shared__ float sVal[8][16];
  __shared__ int   sIdx[8][16];

  const int tid   = threadIdx.x;
  const int lane  = tid & 31;
  const int wave  = tid >> 5;
  const int l16   = lane & 15;
  const int hhalf = lane >> 4;        // 0: lanes 0-15, 1: lanes 16-31
  const int mtile = wave >> 1;        // 0..3
  const int nhalf = wave & 1;         // 0 or 1
  const int nbase = nhalf * (NTOK / 2);

  // ---- Load this wave's 16-row A tile and split into bf16 hi/lo -----------
  // A 16-bit 16x32 layout: lanes<16 hold K = {kb+0..7, kb+16..23},
  //                        lanes>=16 hold K = {kb+8..15, kb+24..31}.
  const int rowbase = blockIdx.x * 64 + mtile * 16;
  const float* xrow = X + (size_t)(rowbase + l16) * EMB;
  v16bf ahi[2], alo[2];
  #pragma unroll
  for (int t = 0; t < 2; ++t) {
    const int kb = t * 32;
    #pragma unroll
    for (int j = 0; j < 8; ++j) {
      float v0 = xrow[kb + 8 * hhalf + j];
      float v1 = xrow[kb + 16 + 8 * hhalf + j];
      __bf16 h0 = (__bf16)v0, h1 = (__bf16)v1;
      ahi[t][j]     = h0;  alo[t][j]     = (__bf16)(v0 - (float)h0);
      ahi[t][8 + j] = h1;  alo[t][8 + j] = (__bf16)(v1 - (float)h1);
    }
  }

  // ---- Staging assignment: each thread moves 32B of the 8KB B buffer ------
  const int sc = tid >> 6;                 // chunk 0..3
  const int so = (tid & 63) * 16;          // bf16 offset within chunk (32B)
  const __bf16* splane = (sc & 1) ? wlo : whi;
  const size_t  sgbase = (size_t)((sc >> 1) ? (NTOK / 2) : 0) * EMB + so;
  __bf16* sdst = &sB[sc][0][0] + so;

  // B 16-bit 32x16 layout: lanes<16 hold K={kb+0..7, kb+16..23}, lanes>=16 K={kb+8..15, kb+24..31},
  // column N = n0 + l16. Two 8-runs per lane per kb.
  typedef __attribute__((ext_vector_type(8))) __bf16 v8bf;
  const __bf16* bh = &sB[nhalf * 2 + 0][l16][0] + 8 * hhalf;
  const __bf16* bl = &sB[nhalf * 2 + 1][l16][0] + 8 * hhalf;

  float bestv[8];
  int   besti[8];
  #pragma unroll
  for (int r = 0; r < 8; ++r) { bestv[r] = 3.402823e38f; besti[r] = 0; }

  // Prefetch iteration 0
  const v4f* gsrc = (const v4f*)(splane + sgbase);
  v4f p0 = gsrc[0], p1 = gsrc[1];

  for (int i = 0; i < NHALF; ++i) {
    __syncthreads();                       // prior iteration done reading sB
    *(v4f*)(sdst)     = p0;
    *(v4f*)(sdst + 8) = p1;
    if (i + 1 < NHALF) {                   // prefetch next tile (overlaps WMMA)
      const v4f* g = (const v4f*)(splane + sgbase + (size_t)(i + 1) * 16 * EMB);
      p0 = g[0]; p1 = g[1];
    }
    __syncthreads();                       // staged tile visible

    v16bf bh0 = __builtin_shufflevector(*(const v8bf*)(bh), *(const v8bf*)(bh + 16), 0,1,2,3,4,5,6,7,8,9,10,11,12,13,14,15);
    v16bf bh1 = __builtin_shufflevector(*(const v8bf*)(bh + 32), *(const v8bf*)(bh + 48), 0,1,2,3,4,5,6,7,8,9,10,11,12,13,14,15);
    v16bf bl0 = __builtin_shufflevector(*(const v8bf*)(bl), *(const v8bf*)(bl + 16), 0,1,2,3,4,5,6,7,8,9,10,11,12,13,14,15);
    v16bf bl1 = __builtin_shufflevector(*(const v8bf*)(bl + 32), *(const v8bf*)(bl + 48), 0,1,2,3,4,5,6,7,8,9,10,11,12,13,14,15);

    // Three independent 2-deep accumulator chains (interleaved issue):
    //   c0 = hi.hi, c1 = hi.lo, c2 = lo.hi
    v8f c0 = {}, c1 = {}, c2 = {};
    c0 = __builtin_amdgcn_wmma_f32_16x16x32_bf16(false, ahi[0], false, bh0, (short)0, c0, false, false);
    c1 = __builtin_amdgcn_wmma_f32_16x16x32_bf16(false, ahi[0], false, bl0, (short)0, c1, false, false);
    c2 = __builtin_amdgcn_wmma_f32_16x16x32_bf16(false, alo[0], false, bh0, (short)0, c2, false, false);
    c0 = __builtin_amdgcn_wmma_f32_16x16x32_bf16(false, ahi[1], false, bh1, (short)0, c0, false, false);
    c1 = __builtin_amdgcn_wmma_f32_16x16x32_bf16(false, ahi[1], false, bl1, (short)0, c1, false, false);
    c2 = __builtin_amdgcn_wmma_f32_16x16x32_bf16(false, alo[1], false, bh1, (short)0, c2, false, false);

    const int n  = nbase + i * 16 + l16;
    const float wq = wsq[n];
    #pragma unroll
    for (int r = 0; r < 8; ++r) {
      // ||w||^2 - 2*(c0+c1+c2) as a pure FMA chain (no extra adds)
      float s = fmaf(-2.0f, c0[r], fmaf(-2.0f, c1[r], fmaf(-2.0f, c2[r], wq)));
      if (s < bestv[r]) { bestv[r] = s; besti[r] = n; }
    }
  }

  // ---- Butterfly min-reduce across the 16 lanes holding each row ----------
  #pragma unroll
  for (int r = 0; r < 8; ++r) {
    float v = bestv[r];
    int   ix = besti[r];
    #pragma unroll
    for (int off = 1; off < 16; off <<= 1) {
      float ov = __shfl_xor(v, off, 32);
      int   oi = __shfl_xor(ix, off, 32);
      if (ov < v || (ov == v && oi < ix)) { v = ov; ix = oi; }
    }
    if (l16 == 0) { sVal[wave][r + 8 * hhalf] = v; sIdx[wave][r + 8 * hhalf] = ix; }
  }
  __syncthreads();

  // ---- Combine the two N-half waves per M-tile, emit index as float -------
  if (tid < 64) {
    const int mt = tid >> 4, rr = tid & 15;
    float v0 = sVal[2 * mt][rr],     v1 = sVal[2 * mt + 1][rr];
    int   i0 = sIdx[2 * mt][rr],     i1 = sIdx[2 * mt + 1][rr];
    int best = (v1 < v0 || (v1 == v0 && i1 < i0)) ? i1 : i0;
    out[(size_t)blockIdx.x * 64 + tid] = (float)best;
  }
}

extern "C" void kernel_launch(void* const* d_in, const int* in_sizes, int n_in,
                              void* d_out, int out_size, void* d_ws, size_t ws_size,
                              hipStream_t stream) {
  const float* X = (const float*)d_in[0];       // [8,2048,64] f32
  const float* W = (const float*)d_in[1];       // [8192,64]  f32
  float* out = (float*)d_out;                   // [16384] indices as float

  __bf16* whi = (__bf16*)d_ws;                  // 1 MB
  __bf16* wlo = whi + (size_t)NTOK * EMB;       // 1 MB
  float*  wsq = (float*)(wlo + (size_t)NTOK * EMB);  // 32 KB

  vq_prep_kernel<<<NTOK / 8, 256, 0, stream>>>(W, whi, wlo, wsq);
  vq_argmin_kernel<<<MROWS / 64, 256, 0, stream>>>(X, whi, wlo, wsq, out);
}
